// PTQLinear_33182917329332
// MI455X (gfx1250) — matrix operand order repaired and checked by reading, compiled, not run on hardware
//
#include <hip/hip_runtime.h>

// GEMM: out[M,N] = x[M,K] * W[N,K]^T * scale + bias,  W = (q_int - zp)
//   M = 32*8 = 256, K = 4096, N = 11008
// Memory-bound on the int32 weight stream (~180 MB) -> read once with B128
// loads, convert int32 -> f16 in-register (codes are exact in f16), run
// v_wmma_f32_16x16x32_f16 from double-buffered LDS, f32 epilogue applies
// scale and bias.

typedef _Float16 v16h __attribute__((ext_vector_type(16)));
typedef _Float16 h8   __attribute__((ext_vector_type(8)));
typedef float    v8f  __attribute__((ext_vector_type(8)));

#define M_TOT 256
#define N_TOT 11008
#define K_TOT 4096
#define BM    128
#define BN    128
#define KB    32
#define LDH   40                 // halfs per LDS row: 32 data + 8 pad = 80 B (16B aligned, conflict-free)
#define NKT   (K_TOT / KB)       // 128 K-steps

__device__ __forceinline__ void load_tile(const float* __restrict__ aG,
                                          const int* __restrict__ bG,
                                          int kt, float4 a4[4], int4 b4[4]) {
  const float4* ap = (const float4*)(aG + (size_t)kt * KB);
  const int4*   bp = (const int4*)(bG + (size_t)kt * KB);
#pragma unroll
  for (int i = 0; i < 4; ++i) { a4[i] = ap[i]; b4[i] = bp[i]; }
}

__device__ __forceinline__ void store_tile(_Float16* aS, _Float16* bS,
                                           const float4 a4[4], const int4 b4[4],
                                           float zp) {
  const float* af = (const float*)a4;
  const int*   bi = (const int*)b4;
#pragma unroll
  for (int c = 0; c < 2; ++c) {
    h8 v;
#pragma unroll
    for (int i = 0; i < 8; ++i) v[i] = (_Float16)af[c * 8 + i];
    *(h8*)(aS + c * 8) = v;
  }
#pragma unroll
  for (int c = 0; c < 2; ++c) {
    h8 v;
#pragma unroll
    for (int i = 0; i < 8; ++i) v[i] = (_Float16)((float)bi[c * 8 + i] - zp);
    *(h8*)(bS + c * 8) = v;
  }
}

__global__ __launch_bounds__(256) void ptq_linear_wmma_f16(
    const float* __restrict__ x,       // [M_TOT, K_TOT]
    const int*   __restrict__ q,       // [N_TOT, K_TOT] int8 codes in int32
    const float* __restrict__ scale_p, // scalar
    const float* __restrict__ zp_p,    // scalar
    const float* __restrict__ bias,    // [N_TOT]
    float*       __restrict__ out)     // [M_TOT, N_TOT]
{
  __shared__ _Float16 As[2][BM][LDH];  // 2 * 128 * 80 B = 20 KB
  __shared__ _Float16 Bs[2][BN][LDH];  // 20 KB

  const int t    = threadIdx.x;
  const int lane = t & 31;
  const int wave = t >> 5;
  const int wm   = wave & 1;   // 2 waves along M -> 64 rows each
  const int wn   = wave >> 1;  // 4 waves along N -> 32 cols each
  const int half = lane >> 4;  // 0: lanes 0-15, 1: lanes 16-31
  const int l16  = lane & 15;

  const int m0 = blockIdx.y * BM;
  const int n0 = blockIdx.x * BN;

  const float zp = zp_p[0];
  const float sc = scale_p[0];

  // Global staging: thread t owns 16 contiguous elements of one row.
  const int srow = t >> 1;
  const int scol = (t & 1) << 4;
  const float* aG = x + (size_t)(m0 + srow) * K_TOT + scol;
  const int*   bG = q + (size_t)(n0 + srow) * K_TOT + scol;

  v8f acc[4][2];
#pragma unroll
  for (int mf = 0; mf < 4; ++mf)
#pragma unroll
    for (int nf = 0; nf < 2; ++nf) acc[mf][nf] = (v8f){0,0,0,0,0,0,0,0};

  float4 a4[4];
  int4   b4[4];

  // Prologue: stage K-step 0.
  load_tile(aG, bG, 0, a4, b4);
  store_tile(&As[0][srow][scol], &Bs[0][srow][scol], a4, b4, zp);
  __syncthreads();

  for (int kt = 0; kt < NKT; ++kt) {
    const int cur = kt & 1;

    if (kt + 1 < NKT) load_tile(aG, bG, kt + 1, a4, b4);
    if (kt + 2 < NKT) {
      __builtin_prefetch(aG + (size_t)(kt + 2) * KB, 0, 0);
      __builtin_prefetch(bG + (size_t)(kt + 2) * KB, 0, 0);
    }

    // Fragment loads from LDS per documented CDNA5 wave32 layouts.
    // A 16x32 f16: lanes 0-15 row M=lane K in {0..7,16..23}; lanes 16-31 K in {8..15,24..31}.
    v16h afr[4];
#pragma unroll
    for (int mf = 0; mf < 4; ++mf) {
      const int row = wm * 64 + mf * 16 + l16;
      h8 c0 = *(const h8*)&As[cur][row][8 * half];        // K = 8*half + 0..7
      h8 c1 = *(const h8*)&As[cur][row][16 + 8 * half];   // K = 16 + 8*half + 0..7
      afr[mf] = __builtin_shufflevector(c0, c1, 0,1,2,3,4,5,6,7,8,9,10,11,12,13,14,15);
    }
    // B 32x16 f16: lanes 0-15 col N=lane K in {0..7,16..23}; lanes 16-31 K in {8..15,24..31}.
    v16h bfr[2];
#pragma unroll
    for (int nf = 0; nf < 2; ++nf) {
      const int row = wn * 32 + nf * 16 + l16;            // W row == output col n
      h8 c0 = *(const h8*)&Bs[cur][row][8 * half];
      h8 c1 = *(const h8*)&Bs[cur][row][16 + 8 * half];
      bfr[nf] = __builtin_shufflevector(c0, c1, 0,1,2,3,4,5,6,7,8,9,10,11,12,13,14,15);
    }

#pragma unroll
    for (int mf = 0; mf < 4; ++mf)
#pragma unroll
      for (int nf = 0; nf < 2; ++nf)
        acc[mf][nf] = __builtin_amdgcn_wmma_f32_16x16x32_f16(
            /*neg_a=*/false, afr[mf], /*neg_b=*/false, bfr[nf],
            /*c_mod=*/(short)0, acc[mf][nf], /*reuse_a=*/false, /*reuse_b=*/false);

    if (kt + 1 < NKT)
      store_tile(&As[cur ^ 1][srow][scol], &Bs[cur ^ 1][srow][scol], a4, b4, zp);
    __syncthreads();
  }

  // Epilogue: D layout -> VGPR r: lanes 0-15 hold M=r, lanes 16-31 hold M=r+8; N = lane%16.
#pragma unroll
  for (int mf = 0; mf < 4; ++mf) {
#pragma unroll
    for (int nf = 0; nf < 2; ++nf) {
      const int nn = n0 + wn * 32 + nf * 16 + l16;
      const float b = bias[nn];
#pragma unroll
      for (int r = 0; r < 8; ++r) {
        const int m = m0 + wm * 64 + mf * 16 + half * 8 + r;
        out[(size_t)m * N_TOT + nn] = acc[mf][nf][r] * sc + b;
      }
    }
  }
}

extern "C" void kernel_launch(void* const* d_in, const int* in_sizes, int n_in,
                              void* d_out, int out_size, void* d_ws, size_t ws_size,
                              hipStream_t stream) {
  (void)in_sizes; (void)n_in; (void)out_size; (void)d_ws; (void)ws_size;
  const float* x     = (const float*)d_in[0];
  const int*   q     = (const int*)d_in[1];
  const float* scale = (const float*)d_in[2];
  const float* zp    = (const float*)d_in[3];
  const float* bias  = (const float*)d_in[4];
  float* out = (float*)d_out;

  dim3 grid(N_TOT / BN, M_TOT / BM);   // 86 x 2 workgroups
  dim3 block(256);                     // 8 wave32s
  ptq_linear_wmma_f16<<<grid, block, 0, stream>>>(x, q, scale, zp, bias, out);
}
